// CRFLayer_21242908246695
// MI455X (gfx1250) — matrix operand order repaired and checked by reading, not compiled, not run on hardware
//
#include <hip/hip_runtime.h>
#include <hip/hip_bf16.h>

typedef __attribute__((ext_vector_type(16))) _Float16 v16h;
typedef __attribute__((ext_vector_type(8)))  _Float16 v8h;
typedef __attribute__((ext_vector_type(8)))  float    v8f;

#define NUM_LABELS 64
#define SEQ_LEN    512
#define BATCH      1024
#define LOG2E      1.44269504088896340736f
#define LN2        0.69314718055994530942f

// One wave (32 lanes) owns a 16-row batch tile for all 512 time steps.
// Labels-next (j) live on the WMMA M dimension (4 tiles of 16).
// Batch rows live on the WMMA N dimension (lane % 16).
// A (constant) = exp(T)^T in f16, held in registers for the whole loop.
// B (per step) = exp(score_hat) in f16, rebuilt via wave-private LDS.
__global__ __launch_bounds__(128) void crf_forward_wmma(
    const float* __restrict__ emissions,      // [B, S, L]
    const int*   __restrict__ mask,           // [B, S]
    const float* __restrict__ transitions,    // [L, L]
    const float* __restrict__ start_t,        // [L]
    const float* __restrict__ end_t,          // [L]
    float*       __restrict__ out)            // [B]
{
    __shared__ _Float16 lds[4][16][NUM_LABELS];   // 4 waves/block, 2KB each

    const int lane = threadIdx.x & 31;
    const int n    = lane & 15;      // batch column within tile / A row
    const int h    = lane >> 4;      // half-wave id
    const int wib  = threadIdx.x >> 5;
    const int wid  = blockIdx.x * 4 + wib;
    const int b0   = wid * 16;

    const float* erow = emissions + (size_t)(b0 + n) * SEQ_LEN * NUM_LABELS;
    const int*   mrow = mask      + (size_t)(b0 + n) * SEQ_LEN;

    // ---- Build constant A matrices: A[jt][kh][m=n, K] = exp(T[i, 16*jt+n]) ----
    // 16-bit A layout (16x32): lane m (0..15) both halves hold row m.
    // this code packs contiguous K (B matches): half0 elems 0..15 -> K=0..15 ; half1: K=16..31.
    v16h Amat[4][2];
    #pragma unroll
    for (int jt = 0; jt < 4; ++jt) {
        #pragma unroll
        for (int kh = 0; kh < 2; ++kh) {
            v16h a;
            #pragma unroll
            for (int v = 0; v < 8; ++v) {
                #pragma unroll
                for (int p = 0; p < 2; ++p) {
                    int K = 2 * v + p + 16 * h;
                    int i = kh * 32 + K;                       // previous label
                    float tv = transitions[i * NUM_LABELS + 16 * jt + n];
                    a[2 * v + p] = (_Float16)__builtin_amdgcn_exp2f(tv * LOG2E);
                }
            }
            Amat[jt][kh] = a;
        }
    }

    // ---- State: score_hat (base-2, max-normalized, <= 0) + running log-norm ----
    // hs[jt][r] corresponds to label 16*jt + 8*h + r of batch row b0+n.
    float hs[4][8];
    float M2;
    {
        float m = -3.0e38f;
        #pragma unroll
        for (int jt = 0; jt < 4; ++jt) {
            const float* p0 = erow + 16 * jt + 8 * h;          // t = 0
            float4 ea = *(const float4*)(p0);
            float4 eb = *(const float4*)(p0 + 4);
            float4 sa = *(const float4*)(start_t + 16 * jt + 8 * h);
            float4 sb = *(const float4*)(start_t + 16 * jt + 8 * h + 4);
            float ev[8] = {ea.x, ea.y, ea.z, ea.w, eb.x, eb.y, eb.z, eb.w};
            float sv[8] = {sa.x, sa.y, sa.z, sa.w, sb.x, sb.y, sb.z, sb.w};
            #pragma unroll
            for (int r = 0; r < 8; ++r) {
                float x = (ev[r] + sv[r]) * LOG2E;
                hs[jt][r] = x;
                m = fmaxf(m, x);
            }
        }
        m = fmaxf(m, __shfl_xor(m, 16));
        M2 = m;
        #pragma unroll
        for (int jt = 0; jt < 4; ++jt)
            #pragma unroll
            for (int r = 0; r < 8; ++r) hs[jt][r] -= m;
    }

    auto loadE = [&](int t, float (&e)[4][8], int& mk) {
        const float* p = erow + (size_t)t * NUM_LABELS;
        #pragma unroll
        for (int jt = 0; jt < 4; ++jt) {
            float4 a = *(const float4*)(p + 16 * jt + 8 * h);
            float4 b = *(const float4*)(p + 16 * jt + 8 * h + 4);
            e[jt][0] = a.x; e[jt][1] = a.y; e[jt][2] = a.z; e[jt][3] = a.w;
            e[jt][4] = b.x; e[jt][5] = b.y; e[jt][6] = b.z; e[jt][7] = b.w;
        }
        mk = mrow[t];
        if (t + 2 < SEQ_LEN)   // pull HBM lines 2 steps ahead -> global_prefetch_b8
            __builtin_prefetch(p + 2 * NUM_LABELS, 0, 0);
    };

    auto step = [&](const float (&e)[4][8], int mk) {
        // 1) p = exp2(score_hat) -> f16, scatter into wave-private LDS [b][i]
        #pragma unroll
        for (int jt = 0; jt < 4; ++jt) {
            v8h pk;
            #pragma unroll
            for (int r = 0; r < 8; ++r)
                pk[r] = (_Float16)__builtin_amdgcn_exp2f(hs[jt][r]);
            *(v8h*)&lds[wib][n][16 * jt + 8 * h] = pk;   // ds_store_b128
        }
        // 2) B operands (32x16 f16): lane (n,h) holds K = kh*32 + 16h + 0..15
        //    (same-wave LDS ops are in-order: no barrier needed)
        v16h B0 = *(const v16h*)&lds[wib][n][16 * h];
        v16h B1 = *(const v16h*)&lds[wib][n][32 + 16 * h];
        // 3) GEMM + 4) u = log2(S) + emit*log2e ; 5) row max
        float m = -3.0e38f;
        float u[4][8];
        #pragma unroll
        for (int jt = 0; jt < 4; ++jt) {
            v8f c = {};
            c = __builtin_amdgcn_wmma_f32_16x16x32_f16(false, Amat[jt][0],
                    false, B0, (short)0, c, false, false);
            c = __builtin_amdgcn_wmma_f32_16x16x32_f16(false, Amat[jt][1],
                    false, B1, (short)0, c, false, false);
            #pragma unroll
            for (int r = 0; r < 8; ++r) {
                float uu = fmaf(e[jt][r], LOG2E, __builtin_amdgcn_logf(c[r]));
                u[jt][r] = uu;
                m = fmaxf(m, uu);
            }
        }
        m = fmaxf(m, __shfl_xor(m, 16));
        // 6) masked update of normalized state + running norm
        const bool f = (mk != 0);
        #pragma unroll
        for (int jt = 0; jt < 4; ++jt)
            #pragma unroll
            for (int r = 0; r < 8; ++r)
                hs[jt][r] = f ? (u[jt][r] - m) : hs[jt][r];
        M2 += f ? m : 0.0f;
    };

    // ---- Software-pipelined time loop: emissions double-buffered in VGPRs ----
    float eA[4][8], eB[4][8];
    int mA = 1, mB = 1;
    loadE(1, eA, mA);
    for (int t = 1; t + 1 < SEQ_LEN; t += 2) {
        loadE(t + 1, eB, mB);
        step(eA, mA);
        if (t + 2 < SEQ_LEN) loadE(t + 2, eA, mA);
        step(eB, mB);
    }
    step(eA, mA);   // t = 511 (loaded in the final pipelined loadE)

    // ---- Finale: out[b] = ln2 * (M2 + log2( sum_j exp2(hs + end*log2e) )) ----
    float m2 = -3.0e38f;
    float w[4][8];
    #pragma unroll
    for (int jt = 0; jt < 4; ++jt) {
        float4 a = *(const float4*)(end_t + 16 * jt + 8 * h);
        float4 b = *(const float4*)(end_t + 16 * jt + 8 * h + 4);
        float ev[8] = {a.x, a.y, a.z, a.w, b.x, b.y, b.z, b.w};
        #pragma unroll
        for (int r = 0; r < 8; ++r) {
            float x = fmaf(ev[r], LOG2E, hs[jt][r]);
            w[jt][r] = x;
            m2 = fmaxf(m2, x);
        }
    }
    m2 = fmaxf(m2, __shfl_xor(m2, 16));
    float s = 0.0f;
    #pragma unroll
    for (int jt = 0; jt < 4; ++jt)
        #pragma unroll
        for (int r = 0; r < 8; ++r)
            s += __builtin_amdgcn_exp2f(w[jt][r] - m2);
    s += __shfl_xor(s, 16);
    float res = (M2 + m2 + __builtin_amdgcn_logf(s)) * LN2;
    if (h == 0) out[b0 + n] = res;
}

extern "C" void kernel_launch(void* const* d_in, const int* in_sizes, int n_in,
                              void* d_out, int out_size, void* d_ws, size_t ws_size,
                              hipStream_t stream) {
    const float* emissions   = (const float*)d_in[0];
    const int*   mask        = (const int*)  d_in[1];
    const float* transitions = (const float*)d_in[2];
    const float* start_t     = (const float*)d_in[3];
    const float* end_t       = (const float*)d_in[4];
    float*       out         = (float*)d_out;
    (void)in_sizes; (void)n_in; (void)out_size; (void)d_ws; (void)ws_size;

    // 64 waves total: 16 blocks x 128 threads (4 waves/block, 1 batch-tile/wave)
    dim3 grid(BATCH / 16 / 4);
    dim3 block(128);
    hipLaunchKernelGGL(crf_forward_wmma, grid, block, 0, stream,
                       emissions, mask, transitions, start_t, end_t, out);
}
